// EnhancedSpatioTemporalLayer_58720792870927
// MI455X (gfx1250) — matrix operand order repaired and checked by reading, not compiled, not run on hardware
//
#include <hip/hip_runtime.h>
#include <hip/hip_bf16.h>

// ---- problem dims ----
#define BB   16
#define FIN  3
#define NV   170   // vertices
#define TT   288   // timesteps
#define HH   64    // hidden
#define OO   64    // out features
#define KK   3     // Chebyshev order
#define NPAD 176   // vertex rows padded to 11 tiles of 16
#define MPAD 192   // K-dim (vertices) padded to 6 steps of 32

typedef __attribute__((ext_vector_type(16))) _Float16 v16h;
typedef __attribute__((ext_vector_type(8)))  _Float16 v8h;
typedef __attribute__((ext_vector_type(8)))  float    v8f;

// ---------------- prepack kernels (tiny, run once per launch) ----------------
// chebp[k][n(176)][m(192)] f16, zero padded -> WMMA A operand, row-major rows.
__global__ void prepack_cheb_kernel(const float* __restrict__ cheb,
                                    _Float16* __restrict__ chebp) {
    int i = blockIdx.x * 256 + threadIdx.x;
    if (i >= KK * NPAD * MPAD) return;
    int m = i % MPAD;
    int r = i / MPAD;
    int n = r % NPAD;
    int k = r / NPAD;
    float v = 0.f;
    if (n < NV && m < NV) v = cheb[(k * NV + n) * NV + m];
    chebp[i] = (_Float16)v;
}

// thetaT[k][o(64)][c(64)] f16 = theta[k][c][o] -> WMMA B operand ([N][K] store).
__global__ void prepack_theta_kernel(const float* __restrict__ theta,
                                     _Float16* __restrict__ thetaT) {
    int i = blockIdx.x * 256 + threadIdx.x;
    if (i >= KK * OO * HH) return;
    int c = i % HH;
    int r = i / HH;
    int o = r % OO;
    int k = r / OO;
    thetaT[i] = (_Float16)theta[(k * HH + c) * OO + o];
}

__device__ __forceinline__ v16h load_a_frag(const _Float16* p) {
    return *(const v16h*)p;
}

// ---------------- fused main kernel: one block per (b, tau) ----------------
__global__ void __launch_bounds__(256)
stgcn_fused_kernel(const float* __restrict__ x,
                   const float* __restrict__ conv_w,
                   const float* __restrict__ conv_b,
                   const float* __restrict__ res_w,
                   const float* __restrict__ res_b,
                   const _Float16* __restrict__ chebp,
                   const _Float16* __restrict__ thetaT,
                   float* __restrict__ out) {
    __shared__ _Float16 t_lds[MPAD * HH];     // [m][h]  24576 B
    __shared__ _Float16 vT[OO * MPAD];        // [o][m]  24576 B (one k at a time)
    __shared__ float    xs[FIN * MPAD * 4];   // [f][m][tap0..2,pad]  9216 B

    const int tid  = threadIdx.x;
    const int tau  = blockIdx.x;     // tau-major: neighboring blocks share out cachelines
    const int b    = blockIdx.y;
    const int lane = tid & 31;
    const int lg   = lane >> 4;      // half-wave group (0/1)
    const int ln   = lane & 15;
    // wave id as a compiler-visible scalar -> SGPR loop control, no EXEC masking
    const int wvu   = __builtin_amdgcn_readfirstlane(tid >> 5);
    const int ot    = wvu & 3;       // this wave's fixed O-column tile
    const int base2 = wvu >> 2;      // row-tile parity (0/1)

    // ---- Stage A: stage x taps (tau-1, tau, tau+1) into LDS ----
    for (int i = tid; i < FIN * MPAD; i += 256) {
        int f = i / MPAD, m = i % MPAD;
        float v0 = 0.f, v1 = 0.f, v2 = 0.f;
        if (m < NV) {
            const float* xp = x + ((size_t)(b * FIN + f) * NV + m) * TT + tau;
            if (tau > 0)      v0 = xp[-1];
            v1 = xp[0];
            if (tau < TT - 1) v2 = xp[1];
        }
        float* q = xs + i * 4;
        q[0] = v0; q[1] = v1; q[2] = v2; q[3] = 0.f;
    }

    // per-thread fixed output channel h for the temporal conv
    const int h = tid & 63;
    float wgt[9];
#pragma unroll
    for (int j = 0; j < 9; ++j) wgt[j] = conv_w[h * 9 + j];
    const float bias = conv_b[h];
    __syncthreads();

    // ---- Stage B: temporal conv + ReLU -> t_lds[m][h] (f16), zero pad m>=170 ----
    for (int it = 0; it < MPAD / 4; ++it) {
        int m = (tid >> 6) + 4 * it;
        float acc = 0.f;
        if (m < NV) {
            acc = bias;
#pragma unroll
            for (int f = 0; f < FIN; ++f) {
                const float* q = xs + (f * MPAD + m) * 4;
                acc = fmaf(q[0], wgt[f * 3 + 0], acc);
                acc = fmaf(q[1], wgt[f * 3 + 1], acc);
                acc = fmaf(q[2], wgt[f * 3 + 2], acc);
            }
            acc = fmaxf(acc, 0.f);
        }
        t_lds[m * HH + h] = (_Float16)acc;
    }
    __syncthreads();

    // ---- Stages C/D per Chebyshev order k ----
    v8f accs[6] = {};   // (nt = base2 + 2j, ot) output tiles, f32 accum

    for (int k = 0; k < KK; ++k) {
        // C_k: v = t @ theta_k.  Wave covers mt = base2 + 2j, j=0..5 (all 12 mt).
        // B-fragments (theta) are wave-invariant across mt: hoist once per k.
        const _Float16* ptb =
            thetaT + ((k * OO + ot * 16 + ln) * HH + lg * 16);
        v16h tb0 = *(const v16h*)(ptb);
        v16h tb1 = *(const v16h*)(ptb + 32);
#pragma unroll
        for (int j = 0; j < 6; ++j) {
            int mt = base2 + 2 * j;
            v8f c = {};
            const _Float16* pa = t_lds + (mt * 16 + ln) * HH + lg * 16;
            v16h a0 = load_a_frag(pa);
            v16h a1 = load_a_frag(pa + 32);
            c = __builtin_amdgcn_wmma_f32_16x16x32_f16(
                    false, a0, false, tb0, (short)0, c, false, false);
            c = __builtin_amdgcn_wmma_f32_16x16x32_f16(
                    false, a1, false, tb1, (short)0, c, false, false);
            // C-fragment: lane owns (M = mt*16 + lg*8 + g, N = ot*16 + ln)
            // -> store transposed f16 into vT[o][m]: 8 consecutive m = one b128
            _Float16* dst = vT + (ot * 16 + ln) * MPAD + mt * 16 + lg * 8;
            v8h hv;
#pragma unroll
            for (int g = 0; g < 8; ++g) hv[g] = (_Float16)c[g];
            *(v8h*)dst = hv;
        }
        __syncthreads();

        // D_k: y += cheb_k @ v.  Wave covers nt = base2 + 2j (nt < 11).
        // B-fragments (vT) are wave-invariant across nt: hoist all 6 per k.
        v16h vb[MPAD / 32];
        {
            const _Float16* pvb = vT + (ot * 16 + ln) * MPAD + lg * 16;
#pragma unroll
            for (int ms = 0; ms < MPAD / 32; ++ms)
                vb[ms] = *(const v16h*)(pvb + ms * 32);
        }
#pragma unroll
        for (int j = 0; j < 6; ++j) {
            int nt = base2 + 2 * j;
            if (nt >= 11) break;   // scalar (SGPR) condition
            v8f c = accs[j];
            const _Float16* pa =
                chebp + ((size_t)(k * NPAD + nt * 16 + ln) * MPAD + lg * 16);
#pragma unroll
            for (int ms = 0; ms < MPAD / 32; ++ms) {
                v16h a = load_a_frag(pa + ms * 32);
                c = __builtin_amdgcn_wmma_f32_16x16x32_f16(
                        false, a, false, vb[ms], (short)0, c, false, false);
            }
            accs[j] = c;
        }
        __syncthreads();   // vT reused next k
    }

    // ---- Epilogue: ReLU(gcn) + residual, store to out[B][O][N][T] ----
    const int o = ot * 16 + ln;
    const float rw0 = res_w[o * 3 + 0];
    const float rw1 = res_w[o * 3 + 1];
    const float rw2 = res_w[o * 3 + 2];
    const float rb  = res_b[o];
#pragma unroll
    for (int j = 0; j < 6; ++j) {
        int nt = base2 + 2 * j;
        if (nt >= 11) break;
        v8f c = accs[j];
#pragma unroll
        for (int g = 0; g < 8; ++g) {
            int n = nt * 16 + lg * 8 + g;
            if (n < NV) {
                float r = rb;
                r = fmaf(xs[(0 * MPAD + n) * 4 + 1], rw0, r);
                r = fmaf(xs[(1 * MPAD + n) * 4 + 1], rw1, r);
                r = fmaf(xs[(2 * MPAD + n) * 4 + 1], rw2, r);
                out[(((size_t)b * OO + o) * NV + n) * TT + tau] =
                    fmaxf(c[g], 0.f) + r;
            }
        }
    }
}

extern "C" void kernel_launch(void* const* d_in, const int* in_sizes, int n_in,
                              void* d_out, int out_size, void* d_ws, size_t ws_size,
                              hipStream_t stream) {
    const float* x      = (const float*)d_in[0];
    // d_in[1] = adj (unused by forward)
    const float* cheb   = (const float*)d_in[2];
    const float* conv_w = (const float*)d_in[3];
    const float* conv_b = (const float*)d_in[4];
    const float* theta  = (const float*)d_in[5];
    const float* res_w  = (const float*)d_in[6];
    const float* res_b  = (const float*)d_in[7];
    float* out = (float*)d_out;

    _Float16* chebp  = (_Float16*)d_ws;                       // 3*176*192 f16
    _Float16* thetaT = chebp + (size_t)KK * NPAD * MPAD;      // 3*64*64 f16

    int n_cheb = KK * NPAD * MPAD;
    int n_th   = KK * OO * HH;
    prepack_cheb_kernel<<<(n_cheb + 255) / 256, 256, 0, stream>>>(cheb, chebp);
    prepack_theta_kernel<<<(n_th + 255) / 256, 256, 0, stream>>>(theta, thetaT);

    dim3 grid(TT, BB);   // tau-major for L2-coalesced [B,O,N,T] stores
    stgcn_fused_kernel<<<grid, 256, 0, stream>>>(
        x, conv_w, conv_b, res_w, res_b, chebp, thetaT, out);
}
